// SparseFeatureWeavingLayer_72155450573412
// MI455X (gfx1250) — matrix operand order repaired and checked by reading, not compiled, not run on hardware
//
#include <hip/hip_runtime.h>
#include <hip/hip_bf16.h>

typedef __bf16 bf16_t;
typedef bf16_t v16bf  __attribute__((ext_vector_type(16)));
typedef bf16_t bf16x8 __attribute__((ext_vector_type(8)));
typedef float  v8f    __attribute__((ext_vector_type(8)));
typedef unsigned int  u32x4 __attribute__((ext_vector_type(4)));
typedef int           i32x4 __attribute__((ext_vector_type(4)));
typedef int           i32x8 __attribute__((ext_vector_type(8)));

#define E_EDGES 262144      // 2^18
#define NNODES  512
#define D_IN    32
#define MID     32
#define OUTC    64
#define NB      2           // batch
#define NROWS   (NB*2*E_EDGES)          // 1,048,576 rows, r = (b*2+d)*E + e
#define TOTAL_X (NROWS*64)              // 67,108,864
#define REGION  (NB*E_EDGES*OUTC)       // 33,554,432 = 2^25

// ---- order-preserving float<->uint mapping for atomic max ----
__device__ __forceinline__ unsigned fenc(float f) {
    int i = __float_as_int(f);
    return (i >= 0) ? (unsigned(i) | 0x80000000u) : unsigned(~i);
}
__device__ __forceinline__ float fdec(unsigned u) {
    int i = (u & 0x80000000u) ? int(u & 0x7FFFFFFFu) : ~int(u);
    return __int_as_float(i);
}
__device__ __forceinline__ float fdec0(unsigned u) {   // decode, non-finite -> 0
    float v = fdec(u);
    return (((__float_as_uint(v) >> 23) & 0xFFu) != 0xFFu) ? v : 0.0f;
}

// K1: per-(side,batch) segment sums into LDS (512x32 f32 = 64KB), merged via global atomics
__global__ void k_seg_sum(const float* __restrict__ xs, const int* __restrict__ a2b,
                          const int* __restrict__ b2a, float* __restrict__ gsum,
                          float* __restrict__ gcnt)
{
    __shared__ float ls[NNODES * D_IN];
    const int tid = threadIdx.x;
    for (int i = tid; i < NNODES * D_IN; i += blockDim.x) ls[i] = 0.0f;
    __syncthreads();

    const int sb = blockIdx.y;          // 0..3
    const int s = sb >> 1, b = sb & 1;  // s = side, b = batch
    const int* src = (s == 0) ? (a2b + (b * 2 + 0) * E_EDGES)
                              : (b2a + (b * 2 + 0) * E_EDGES);
    const float* x = xs + ((s * NB + b) * E_EDGES) * D_IN;

    const int CH = E_EDGES / gridDim.x;
    const int e0 = blockIdx.x * CH;
    for (int e = e0 + tid; e < e0 + CH; e += blockDim.x) {
        const int node = src[e];
        const float* xe = x + e * D_IN;
        #pragma unroll
        for (int dim = 0; dim < D_IN; ++dim)
            atomicAdd(&ls[node * D_IN + dim], xe[dim]);
        atomicAdd(&gcnt[(s * NB + b) * NNODES + node], 1.0f);
    }
    __syncthreads();
    float* gs = gsum + (s * NB + b) * NNODES * D_IN;
    for (int i = tid; i < NNODES * D_IN; i += blockDim.x)
        if (ls[i] != 0.0f) atomicAdd(&gs[i], ls[i]);
}

// K2: sum -> mean (in place)
__global__ void k_seg_fin(float* __restrict__ gsum, const float* __restrict__ gcnt)
{
    const int i = blockIdx.x * blockDim.x + threadIdx.x;
    if (i >= 4 * NNODES * D_IN) return;
    const float c = gcnt[i >> 5];
    gsum[i] = gsum[i] / fmaxf(c, 1.0f);
}

// K3: build x = concat([xs[d], gather(mean_other, dst)]) as bf16, layout (B,2,E,64)
__global__ void k_build_x(const float* __restrict__ xs, const float* __restrict__ mean,
                          const int* __restrict__ a2b, const int* __restrict__ b2a,
                          bf16_t* __restrict__ xbf)
{
    const int i = blockIdx.x * blockDim.x + threadIdx.x;
    if (i >= TOTAL_X) return;
    const int c = i & 63;
    const int r = i >> 6;
    const int e = r & (E_EDGES - 1);
    const int d = (r >> 18) & 1;
    const int b = r >> 19;
    float v;
    if (c < D_IN) {
        v = xs[((d * NB + b) * E_EDGES + e) * D_IN + c];
    } else {
        const int dst = (d == 0) ? a2b[(b * 2 + 1) * E_EDGES + e]
                                 : b2a[(b * 2 + 1) * E_EDGES + e];
        const int srcside = 1 - d;
        v = mean[((srcside * NB + b) * NNODES + dst) * D_IN + (c - D_IN)];
    }
    xbf[i] = (bf16_t)v;
}

// load a 16x32 bf16 A fragment (ISA 16-bit A layout) from a row-major 64-col matrix
// lane holds row = lane&15; two 8-element runs: k0 = ks*32 + hi*8, k1 = ks*32 + 16 + hi*8
__device__ __forceinline__ v16bf load_a_frag(const bf16x8* row8, int ks, int hi) {
    const bf16x8 lo = row8[ks * 4 + hi];
    const bf16x8 up = row8[ks * 4 + 2 + hi];
    return __builtin_shufflevector(lo, up, 0,1,2,3,4,5,6,7,8,9,10,11,12,13,14,15);
}

// K4: z = x @ Wmax^T + b_max via WMMA bf16, fused segment-max scatter (mapped-uint atomics)
__global__ void k_gemm1_segmax(const bf16_t* __restrict__ xbf, const float* __restrict__ Wmax,
                               const float* __restrict__ bmax, const int* __restrict__ a2b,
                               const int* __restrict__ b2a, unsigned* __restrict__ zmaxu)
{
    __shared__ bf16_t sW[MID * 64];     // 32x64 bf16 = 4 KB
    const int tid = threadIdx.x;
    for (int i = tid; i < MID * 64; i += blockDim.x) sW[i] = (bf16_t)Wmax[i];
    __syncthreads();

    const int lane = tid & 31;
    const int tile = blockIdx.x * (blockDim.x >> 5) + (tid >> 5);
    const int row0 = tile * 16;
    const int lrow = lane & 15, hi = lane >> 4;

    // A fragments via 128-bit loads
    const bf16x8* arow8 = (const bf16x8*)(xbf + (row0 + lrow) * 64);
    v16bf a[2];
    a[0] = load_a_frag(arow8, 0, hi);
    a[1] = load_a_frag(arow8, 1, hi);

    // B fragments: element(k=c, n=m) = Wmax[m*64 + c]; two 8-element runs per lane (k = hi*8.., 16+hi*8..)
    v16bf bf[2][2];
    #pragma unroll
    for (int nt = 0; nt < 2; ++nt)
        #pragma unroll
        for (int ks = 0; ks < 2; ++ks) {
            const bf16x8* wrow8 = (const bf16x8*)(sW + (nt * 16 + lrow) * 64 + ks * 32 + hi * 8);
            bf[nt][ks] = __builtin_shufflevector(wrow8[0], wrow8[2],
                                                 0,1,2,3,4,5,6,7,8,9,10,11,12,13,14,15);
        }

    v8f acc[2] = {};
    #pragma unroll
    for (int nt = 0; nt < 2; ++nt) {
        acc[nt] = __builtin_amdgcn_wmma_f32_16x16x32_bf16(false, a[0], false, bf[nt][0],
                                                          (short)0, acc[nt], false, false);
        acc[nt] = __builtin_amdgcn_wmma_f32_16x16x32_bf16(false, a[1], false, bf[nt][1],
                                                          (short)0, acc[nt], false, false);
    }

    const float bb0 = bmax[lrow], bb1 = bmax[16 + lrow];

    // segment-max scatter: acc element i is row (row0 + hi*8 + i), col (nt*16 + lrow)
    int zbase[8];
    #pragma unroll
    for (int i = 0; i < 8; ++i) {
        const int r = row0 + hi * 8 + i;
        const int e = r & (E_EDGES - 1);
        const int d = (r >> 18) & 1;
        const int b = r >> 19;
        const int node = (d == 0) ? a2b[(b * 2 + 0) * E_EDGES + e]
                                  : b2a[(b * 2 + 0) * E_EDGES + e];
        zbase[i] = ((b * 2 + d) * NNODES + node) * MID;
    }
    #pragma unroll
    for (int i = 0; i < 8; ++i) {
        atomicMax(&zmaxu[zbase[i] + lrow],      fenc(acc[0][i] + bb0));
        atomicMax(&zmaxu[zbase[i] + 16 + lrow], fenc(acc[1][i] + bb1));
    }
}

// K5: h = [x | gather(zmax)] @ Wconv^T + b_conv via WMMA bf16; write pre-BN to d_out (2,B,E,64)
// W_conv staged into LDS by the Tensor Data Mover (one wave issues tensor_load_to_lds).
__global__ void k_gemm2(const bf16_t* __restrict__ xbf, const unsigned* __restrict__ zmaxu,
                        const float* __restrict__ Wconv, const float* __restrict__ bconv,
                        const int* __restrict__ a2b, const int* __restrict__ b2a,
                        float* __restrict__ out)
{
    __shared__ float sWf[OUTC * 96];    // 64x96 f32 = 24 KB, filled by TDM
    const int tid = threadIdx.x;

    if (tid < 32) {
        // ---- Tensor DMA descriptor (D#): 2-D tensor, 4-byte elems, 96x64 tile ----
        const unsigned long long ga = (unsigned long long)(uintptr_t)Wconv;
        const unsigned lds_addr = (unsigned)(uintptr_t)&sWf[0];  // LDS aperture: low 32 bits
        u32x4 g0;
        g0[0] = 1u;                                   // count=1, user mode
        g0[1] = lds_addr;                             // lds_addr[31:0]
        g0[2] = (unsigned)(ga & 0xFFFFFFFFu);         // global_addr[31:0]
        g0[3] = (unsigned)((ga >> 32) & 0x01FFFFFFu)  // global_addr[56:32]
              | (2u << 30);                           // type = 2 (image)
        i32x8 g1;
        g1[0] = (int)(2u << 16);                      // workgroup_mask=0, data_size=2 (4B)
        g1[1] = (int)(96u << 16);                     // tensor_dim0[15:0] = 96
        g1[2] = (int)(64u << 16);                     // tensor_dim0_hi=0, tensor_dim1[15:0]=64
        g1[3] = (int)(96u << 16);                     // tensor_dim1_hi=0, tile_dim0=96
        g1[4] = 64;                                   // tile_dim1=64, tile_dim2=0
        g1[5] = 96;                                   // tensor_dim0_stride[31:0] = 96
        g1[6] = 0;                                    // stride0_hi=0, dim1_stride lo=0
        g1[7] = 0;
        i32x4 gz4 = {0, 0, 0, 0};                     // groups 2/3 unused (<=2D)
        i32x8 gz8 = {0, 0, 0, 0, 0, 0, 0, 0};
        __builtin_amdgcn_tensor_load_to_lds(g0, g1, gz4, gz4, gz8, 0);
        __builtin_amdgcn_s_wait_tensorcnt(0);
    }
    __syncthreads();

    const int lane = tid & 31;
    const int tile = blockIdx.x * (blockDim.x >> 5) + (tid >> 5);
    const int row0 = tile * 16;
    const int lrow = lane & 15, hi = lane >> 4;

    // A fragments 0,1 from x (K = 0..63), 128-bit loads
    const bf16x8* arow8 = (const bf16x8*)(xbf + (row0 + lrow) * 64);
    v16bf a[3];
    a[0] = load_a_frag(arow8, 0, hi);
    a[1] = load_a_frag(arow8, 1, hi);

    // A fragment 2: zg = gather(zmax, end_points), 128-bit loads + decode
    {
        const int r = row0 + lrow;
        const int e = r & (E_EDGES - 1);
        const int d = (r >> 18) & 1;
        const int b = r >> 19;
        const int node = (d == 0) ? a2b[(b * 2 + 0) * E_EDGES + e]
                                  : b2a[(b * 2 + 0) * E_EDGES + e];
        const u32x4* z4 = (const u32x4*)(zmaxu + ((b * 2 + d) * NNODES + node) * MID);
        const u32x4 q0 = z4[hi * 2],     q1 = z4[hi * 2 + 1];      // k = hi*8 .. hi*8+7
        const u32x4 q2 = z4[4 + hi * 2], q3 = z4[5 + hi * 2];      // k = 16+hi*8 ..
        #pragma unroll
        for (int j = 0; j < 4; ++j) {
            a[2][j]      = (bf16_t)fdec0(q0[j]);
            a[2][4 + j]  = (bf16_t)fdec0(q1[j]);
            a[2][8 + j]  = (bf16_t)fdec0(q2[j]);
            a[2][12 + j] = (bf16_t)fdec0(q3[j]);
        }
    }

    v8f acc[4] = {};
    #pragma unroll
    for (int nt = 0; nt < 4; ++nt) {
        #pragma unroll
        for (int ks = 0; ks < 3; ++ks) {
            const float* wrow = sWf + (nt * 16 + lrow) * 96 + ks * 32 + hi * 8;
            v16bf bf;
            #pragma unroll
            for (int j = 0; j < 16; ++j) bf[j] = (bf16_t)wrow[j + (j >> 3) * 8];
            acc[nt] = __builtin_amdgcn_wmma_f32_16x16x32_bf16(false, a[ks], false, bf,
                                                              (short)0, acc[nt], false, false);
        }
    }

    #pragma unroll
    for (int i = 0; i < 8; ++i) {
        const int r = row0 + hi * 8 + i;
        const int e = r & (E_EDGES - 1);
        const int d = (r >> 18) & 1;
        const int b = r >> 19;
        const int obase = ((d * NB + b) * E_EDGES + e) * OUTC;   // swapaxes(0,1) layout
        #pragma unroll
        for (int nt = 0; nt < 4; ++nt) {
            const int o = nt * 16 + lrow;
            out[obase + o] = acc[nt][i] + bconv[o];
        }
    }
}

// K6: per-direction, per-channel sum / sumsq over (B,E)
__global__ void k_bn_stats(const float* __restrict__ out, float* __restrict__ s1,
                           float* __restrict__ s2)
{
    __shared__ float l1[OUTC], l2[OUTC];
    const int d = blockIdx.y;
    if (threadIdx.x < OUTC) { l1[threadIdx.x] = 0.0f; l2[threadIdx.x] = 0.0f; }
    __syncthreads();
    const int per = REGION / gridDim.x;
    const int start = blockIdx.x * per;
    const float* p = out + d * REGION;
    for (int i = start + threadIdx.x; i < start + per; i += blockDim.x) {
        __builtin_prefetch(&p[i + 4096], 0, 0);      // global_prefetch_b8, streaming
        const float v = p[i];
        const int c = i & 63;
        atomicAdd(&l1[c], v);
        atomicAdd(&l2[c], v * v);
    }
    __syncthreads();
    if (threadIdx.x < OUTC) {
        atomicAdd(&s1[d * OUTC + threadIdx.x], l1[threadIdx.x]);
        atomicAdd(&s2[d * OUTC + threadIdx.x], l2[threadIdx.x]);
    }
}

// K7: fold BN stats into per-(direction,channel) scale/shift
__global__ void k_bn_fin(const float* __restrict__ s1, const float* __restrict__ s2,
                         const float* __restrict__ gfw, const float* __restrict__ bfw,
                         const float* __restrict__ gbw, const float* __restrict__ bbw,
                         float* __restrict__ scale, float* __restrict__ shift)
{
    const int t = blockIdx.x * blockDim.x + threadIdx.x;
    if (t >= 2 * OUTC) return;
    const int d = t >> 6, c = t & 63;
    const float N = (float)(NB * E_EDGES);
    const float mean = s1[t] / N;
    const float var  = s2[t] / N - mean * mean;
    const float g  = d ? gbw[c] : gfw[c];
    const float be = d ? bbw[c] : bfw[c];
    const float sc = g * rsqrtf(var + 1e-5f);
    scale[t] = sc;
    shift[t] = be - mean * sc;
}

// K8: in-place BN apply + PReLU
__global__ void k_bn_apply(float* __restrict__ out, const float* __restrict__ scale,
                           const float* __restrict__ shift, const float* __restrict__ pa)
{
    const int i = blockIdx.x * blockDim.x + threadIdx.x;
    if (i >= 2 * REGION) return;
    const int c = i & 63;
    const int d = i >> 25;          // REGION = 2^25
    float v = out[i] * scale[d * OUTC + c] + shift[d * OUTC + c];
    out[i] = (v >= 0.0f) ? v : (*pa) * v;
}

extern "C" void kernel_launch(void* const* d_in, const int* in_sizes, int n_in,
                              void* d_out, int out_size, void* d_ws, size_t ws_size,
                              hipStream_t stream) {
    (void)in_sizes; (void)n_in; (void)out_size; (void)ws_size;
    const float* xs    = (const float*)d_in[0];
    const int*   a2b   = (const int*)  d_in[1];
    const int*   b2a   = (const int*)  d_in[2];
    const float* Wmax  = (const float*)d_in[3];
    const float* bmax  = (const float*)d_in[4];
    const float* Wconv = (const float*)d_in[5];
    const float* bconv = (const float*)d_in[6];
    const float* gfw   = (const float*)d_in[7];
    const float* bfw   = (const float*)d_in[8];
    const float* gbw   = (const float*)d_in[9];
    const float* bbw   = (const float*)d_in[10];
    const float* pa    = (const float*)d_in[11];
    float* out = (float*)d_out;

    // workspace layout (floats)
    float*    wsf   = (float*)d_ws;
    float*    gsum  = wsf;                           // 131072  (means, finalized in place)
    float*    gcnt  = wsf + 131072;                  //   2048
    unsigned* zmaxu = (unsigned*)(wsf + 133120);     //  65536
    float*    bns1  = wsf + 198656;                  //    128
    float*    bns2  = wsf + 198784;                  //    128
    float*    scale = wsf + 198912;                  //    128
    float*    shift = wsf + 199040;                  //    128
    bf16_t*   xbf   = (bf16_t*)((char*)d_ws + 819200);  // 67,108,864 bf16 = 128 MiB

    (void)hipMemsetAsync(d_ws, 0, (size_t)199168 * sizeof(float), stream);

    k_seg_sum <<<dim3(64, 4), 256, 0, stream>>>(xs, a2b, b2a, gsum, gcnt);
    k_seg_fin <<<(4 * NNODES * D_IN + 255) / 256, 256, 0, stream>>>(gsum, gcnt);
    k_build_x <<<TOTAL_X / 256, 256, 0, stream>>>(xs, gsum, a2b, b2a, xbf);

    // 65536 16-row tiles, 4 waves (tiles) per 128-thread block
    k_gemm1_segmax<<<(NROWS / 16) / 4, 128, 0, stream>>>(xbf, Wmax, bmax, a2b, b2a, zmaxu);
    k_gemm2       <<<(NROWS / 16) / 4, 128, 0, stream>>>(xbf, zmaxu, Wconv, bconv, a2b, b2a, out);

    k_bn_stats<<<dim3(2048, 2), 256, 0, stream>>>(out, bns1, bns2);
    k_bn_fin  <<<1, 128, 0, stream>>>(bns1, bns2, gfw, bfw, gbw, bbw, scale, shift);
    k_bn_apply<<<(2 * REGION) / 256, 256, 0, stream>>>(out, scale, shift, pa);
}
